// CausalSelfAttention_27797028340108
// MI455X (gfx1250) — compile-verified, matrix operand order checked by reading, not run on hardware
//
#include <hip/hip_runtime.h>
#include <hip/hip_bf16.h>

typedef __bf16 bf16;
typedef __attribute__((ext_vector_type(16))) __bf16 v16bf;
typedef __attribute__((ext_vector_type(8)))  __bf16 v8bf;
typedef __attribute__((ext_vector_type(8)))  float  v8f;
typedef __attribute__((ext_vector_type(4)))  unsigned u32x4;
typedef __attribute__((ext_vector_type(8)))  int      i32x8;
typedef __attribute__((ext_vector_type(4)))  int      i32x4;

#define BB 4
#define TT 2048
#define CC 1024
#define HH 16
#define DD 64

// ---------------------------------------------------------------------------
// WMMA helper: D = A(16x32 bf16) * B(32x16 bf16) + C(16x16 f32)
// ---------------------------------------------------------------------------
__device__ __forceinline__ v8f wmma_bf16(v16bf a, v16bf b, v8f c) {
    return __builtin_amdgcn_wmma_f32_16x16x32_bf16(
        false, a, false, b, (short)0, c, false, false);
}

// Load one A/B fragment. `p16` points at element (k + 8*(lane>>4)) of the
// lane's row (16B aligned). ISA 7.12.2: VGPR0..3 = K chunk0..+7,
// VGPR4..7 = K chunk0+16..+23.
__device__ __forceinline__ v16bf frag_ld(const bf16* p16) {
    union { v16bf v; v8bf h[2]; } u;
    const v8bf* p = reinterpret_cast<const v8bf*>(p16);
    u.h[0] = p[0];
    u.h[1] = p[2];
    return u.v;
}

// ---------------------------------------------------------------------------
// TDM: DMA a (rows x 32) bf16 tile with 64 rows, row stride TT (2048) elems,
// into LDS (compact, 32 elems/row). D# per CDNA5 ISA §8.3/8.4.
// Wave-uniform; tracked by TENSORcnt. 6-arg builtin (clang-23 signature).
// ---------------------------------------------------------------------------
__device__ __forceinline__ void tdm_load_vt(const bf16* gptr, unsigned lds_byte_off) {
    unsigned long long ga = (unsigned long long)(size_t)gptr;
    u32x4 g0;
    g0[0] = 1u;                                    // count=1 (valid), user mode
    g0[1] = lds_byte_off;                          // lds_addr (bytes)
    g0[2] = (unsigned)(ga & 0xFFFFFFFFu);          // global_addr[31:0]
    g0[3] = (unsigned)((ga >> 32) & 0x1FFFFFFu)    // global_addr[56:32]
          | (2u << 30);                            // type = 2 ("image")
    i32x8 g1;
    g1[0] = 0x10000;        // data_size = 1 (2 bytes/elem)
    g1[1] = TT << 16;       // tensor_dim0 = 2048   (bits 63:48)
    g1[2] = 64 << 16;       // tensor_dim1 = 64     (bits 111:80 low half)
    g1[3] = 32 << 16;       // tile_dim0   = 32     (bits 127:112)
    g1[4] = 64;             // tile_dim1   = 64     (bits 143:128)
    g1[5] = TT;             // tensor_dim0_stride = 2048 (bits 191:160)
    g1[6] = 0;
    g1[7] = 0;
    i32x4 z4;
    z4[0] = 0; z4[1] = 0; z4[2] = 0; z4[3] = 0;
    i32x8 z8;
    z8[0] = 0; z8[1] = 0; z8[2] = 0; z8[3] = 0;
    z8[4] = 0; z8[5] = 0; z8[6] = 0; z8[7] = 0;
    __builtin_amdgcn_tensor_load_to_lds(g0, g1, z4, z4, z8, 0);
}

// ---------------------------------------------------------------------------
// fp32 -> bf16 convert
// ---------------------------------------------------------------------------
__global__ void cvt_f32_bf16(const float* __restrict__ in,
                             bf16* __restrict__ out, int n) {
    int i = blockIdx.x * blockDim.x + threadIdx.x;
    int stride = gridDim.x * blockDim.x;
    for (; i < n; i += stride) out[i] = (bf16)in[i];
}

// ---------------------------------------------------------------------------
// Extract V from qkv[B,T,3C] and write V^T as [B,H,D,T] (keys contiguous).
// Coalesced writes; reads hit L2 (qkv was just produced).
// ---------------------------------------------------------------------------
__global__ void transpose_v(const bf16* __restrict__ qkv, bf16* __restrict__ vt) {
    const long total = (long)BB * CC * TT;          // 8388608
    long i = (long)blockIdx.x * blockDim.x + threadIdx.x;
    const long stride = (long)gridDim.x * blockDim.x;
    for (; i < total; i += stride) {
        const int  t   = (int)(i & (TT - 1));
        const long bhd = i >> 11;                   // b*1024 + h*64 + d
        const int  hd  = (int)(bhd & 1023);         // h*64 + d
        const int  b   = (int)(bhd >> 10);
        vt[i] = qkv[((long)b * TT + t) * (3 * CC) + 2 * CC + hd];
    }
}

// ---------------------------------------------------------------------------
// GEMM: C[M,N] = A[M,K] * W[N,K]^T.  A,W bf16 row-major. One wave = 64x64 tile.
// Block = 256 threads (8 waves): covers 512 rows x 64 cols.
// ---------------------------------------------------------------------------
template <int OUT_BF16>
__global__ void gemm_wmma(const bf16* __restrict__ A,
                          const bf16* __restrict__ W,
                          void* __restrict__ Cout,
                          int M, int N, int K) {
    const int lane  = threadIdx.x & 31;
    const int wave  = threadIdx.x >> 5;
    const int lmod  = lane & 15;
    const int lhalf = lane >> 4;

    const int m0 = blockIdx.y * 512 + wave * 64;
    const int n0 = blockIdx.x * 64;

    v8f acc[4][4];
#pragma unroll
    for (int mt = 0; mt < 4; ++mt)
#pragma unroll
        for (int nt = 0; nt < 4; ++nt)
#pragma unroll
            for (int r = 0; r < 8; ++r) acc[mt][nt][r] = 0.f;

    for (int k = 0; k < K; k += 32) {
        v16bf af[4], bfr[4];
#pragma unroll
        for (int mt = 0; mt < 4; ++mt) {
            const bf16* p = A + (long)(m0 + mt * 16 + lmod) * K + k + 8 * lhalf;
            af[mt] = frag_ld(p);
            __builtin_prefetch(p + 64, 0, 0);      // global_prefetch_b8
        }
#pragma unroll
        for (int nt = 0; nt < 4; ++nt) {
            const bf16* p = W + (long)(n0 + nt * 16 + lmod) * K + k + 8 * lhalf;
            bfr[nt] = frag_ld(p);
            __builtin_prefetch(p + 64, 0, 0);
        }
#pragma unroll
        for (int mt = 0; mt < 4; ++mt)
#pragma unroll
            for (int nt = 0; nt < 4; ++nt)
                acc[mt][nt] = wmma_bf16(af[mt], bfr[nt], acc[mt][nt]);
    }

#pragma unroll
    for (int mt = 0; mt < 4; ++mt)
#pragma unroll
        for (int nt = 0; nt < 4; ++nt)
#pragma unroll
            for (int r = 0; r < 8; ++r) {
                const long row = m0 + mt * 16 + r + 8 * lhalf;
                const long col = n0 + nt * 16 + lmod;
                if (OUT_BF16)
                    ((bf16*)Cout)[row * N + col] = (bf16)acc[mt][nt][r];
                else
                    ((float*)Cout)[row * N + col] = acc[mt][nt][r];
            }
}

// ---------------------------------------------------------------------------
// Flash attention. Q,K from qkv[B,T,3C]; V from vt[B,H,D,T] (pre-transposed).
// One wave = 16-query tile; block = 8 waves = 128 queries of one (b,h).
// Per 32-key block:
//   S = QK^T        : 4 WMMAs (K frags straight from global, b128)
//   softmax         : in-register online update (16-lane shuffles)
//   P               : bounced through LDS to form the A-fragment
//   V^T tile        : TDM DMA into double-buffered LDS (64 d-rows x 32 keys,
//                     keys contiguous) -> B-frags are two ds_load_b128 each
//   O += P*V        : 4 WMMAs
// ---------------------------------------------------------------------------
__global__ void flash_attn(const bf16* __restrict__ qkv,
                           const bf16* __restrict__ vt,
                           bf16* __restrict__ y) {
    const int lane  = threadIdx.x & 31;
    const int wave  = threadIdx.x >> 5;
    const int lmod  = lane & 15;
    const int lhalf = lane >> 4;

    const int bh = blockIdx.y;
    const int b  = bh >> 4;           // H = 16
    const int h  = bh & 15;
    const int q0 = blockIdx.x * 128 + wave * 16;

    // LDS map (element offsets into one array -> addrspace(3) stays visible):
    //   [0, 8*512)            P tiles (16x32 per wave)
    //   [4096, 4096+16*2048)  V^T tiles (2 buffers x (64x32) per wave)
    __shared__ __align__(128) bf16 smem[8 * 512 + 16 * 2048];
    const int pbase  = wave * 512;
    const int vbase0 = 8 * 512 + wave * 2 * 2048;

    const long rowStride = 3 * CC;    // 3072 elements
    const bf16* Qb  = qkv + (long)b * TT * rowStride + 0 * CC + h * DD;
    const bf16* Kb  = qkv + (long)b * TT * rowStride + 1 * CC + h * DD;
    const bf16* Vtb = vt + (long)(b * HH + h) * DD * TT;   // [64][2048]

    // Q fragments, held for the whole key loop (16 rows x 64 D)
    v16bf qf0, qf1;
    {
        const bf16* qrow = Qb + (long)(q0 + lmod) * rowStride + 8 * lhalf;
        qf0 = frag_ld(qrow);
        qf1 = frag_ld(qrow + 32);
    }

    v8f o[4];
    float mi[8], li[8], alpha[8];
#pragma unroll
    for (int nt = 0; nt < 4; ++nt)
#pragma unroll
        for (int r = 0; r < 8; ++r) o[nt][r] = 0.f;
#pragma unroll
    for (int r = 0; r < 8; ++r) { mi[r] = -1e30f; li[r] = 0.f; }

    const float scale = 0.125f;       // 1/sqrt(64)
    const int kend = q0 + 15;
    int cur = 0;

    // Prologue: start DMA of the first V^T tile (keys 0..31).
    tdm_load_vt(Vtb, (unsigned)vbase0 * 2);

    for (int k0 = 0; k0 <= kend; k0 += 32) {
        // ----- S = Q * K^T for 32 keys (two 16-key subtiles) -----
        v8f s0, s1;
#pragma unroll
        for (int r = 0; r < 8; ++r) { s0[r] = 0.f; s1[r] = 0.f; }
        {
            const bf16* kr0 = Kb + (long)(k0 + lmod) * rowStride + 8 * lhalf;
            s0 = wmma_bf16(qf0, frag_ld(kr0),      s0);
            s0 = wmma_bf16(qf1, frag_ld(kr0 + 32), s0);
            const bf16* kr1 = Kb + (long)(k0 + 16 + lmod) * rowStride + 8 * lhalf;
            s1 = wmma_bf16(qf0, frag_ld(kr1),      s1);
            s1 = wmma_bf16(qf1, frag_ld(kr1 + 32), s1);
        }

        // ----- scale, causal mask, online softmax (one fused per-row pass) --
        const int key0 = k0 + lmod, key1 = k0 + 16 + lmod;
        float p0[8], p1[8];
#pragma unroll
        for (int r = 0; r < 8; ++r) {
            const int q = q0 + r + 8 * lhalf;
            float a = s0[r] * scale; if (key0 > q) a = -1e30f;
            float c = s1[r] * scale; if (key1 > q) c = -1e30f;
            float mt = fmaxf(a, c);
            mt = fmaxf(mt, __shfl_xor(mt, 1));
            mt = fmaxf(mt, __shfl_xor(mt, 2));
            mt = fmaxf(mt, __shfl_xor(mt, 4));
            mt = fmaxf(mt, __shfl_xor(mt, 8));
            const float mn = fmaxf(mi[r], mt);
            alpha[r] = __expf(mi[r] - mn);
            mi[r] = mn;
            a = __expf(a - mn);
            c = __expf(c - mn);
            p0[r] = a; p1[r] = c;
            float t = a + c;
            t += __shfl_xor(t, 1);
            t += __shfl_xor(t, 2);
            t += __shfl_xor(t, 4);
            t += __shfl_xor(t, 8);
            li[r] = li[r] * alpha[r] + t;
        }
#pragma unroll
        for (int nt = 0; nt < 4; ++nt)
#pragma unroll
            for (int r = 0; r < 8; ++r) o[nt][r] *= alpha[r];

        // ----- write P to LDS (row-major 16x32) -----
#pragma unroll
        for (int r = 0; r < 8; ++r) {
            const int row = r + 8 * lhalf;
            smem[pbase + row * 32 + lmod]      = (bf16)p0[r];
            smem[pbase + row * 32 + 16 + lmod] = (bf16)p1[r];
        }

        // ----- V^T(cur) ready; kick off DMA of the next tile -----
        const int vb = vbase0 + (cur ? 2048 : 0);
        __builtin_amdgcn_s_wait_tensorcnt(0);
        if (k0 + 32 <= kend) {
            const int vbn = vbase0 + (cur ? 0 : 2048);
            tdm_load_vt(Vtb + (k0 + 32), (unsigned)vbn * 2);
        }
        __builtin_amdgcn_wave_barrier();
        asm volatile("s_wait_dscnt 0" ::: "memory");   // P stores + fence

        // ----- O += P * V : B-frags are two ds_load_b128 each -----
        v16bf pf = frag_ld(&smem[pbase + lmod * 32 + 8 * lhalf]);
#pragma unroll
        for (int nt = 0; nt < 4; ++nt) {
            v16bf vf = frag_ld(&smem[vb + (nt * 16 + lmod) * 32 + 8 * lhalf]);
            o[nt] = wmma_bf16(pf, vf, o[nt]);
        }
        __builtin_amdgcn_wave_barrier();
        cur ^= 1;
    }

    // ----- epilogue: normalize and write y[B,T,C] -----
#pragma unroll
    for (int r = 0; r < 8; ++r) alpha[r] = 1.0f / li[r];
#pragma unroll
    for (int nt = 0; nt < 4; ++nt)
#pragma unroll
        for (int r = 0; r < 8; ++r) {
            const long q = q0 + r + 8 * lhalf;
            const long col = h * DD + nt * 16 + lmod;
            y[((long)b * TT + q) * CC + col] = (bf16)(o[nt][r] * alpha[r]);
        }
}

// ---------------------------------------------------------------------------
// Host launcher
// ---------------------------------------------------------------------------
extern "C" void kernel_launch(void* const* d_in, const int* in_sizes, int n_in,
                              void* d_out, int out_size, void* d_ws, size_t ws_size,
                              hipStream_t stream) {
    const float* x      = (const float*)d_in[0];   // [4,2048,1024]
    const float* w_qkv  = (const float*)d_in[1];   // [3072,1024]
    const float* w_proj = (const float*)d_in[2];   // [1024,1024]

    const int szX  = BB * TT * CC;       // 8388608
    const int szWq = 3 * CC * CC;        // 3145728
    const int szWp = CC * CC;            // 1048576

    char* ws = (char*)d_ws;
    bf16* xb    = (bf16*)(ws);                 // 16 MB
    bf16* wqkvb = (bf16*)(ws + 16777216);      //  6 MB
    bf16* wprjb = (bf16*)(ws + 23068672);      //  2 MB
    bf16* qkvb  = (bf16*)(ws + 25165824);      // 48 MB
    bf16* yb    = (bf16*)(ws + 75497472);      // 16 MB
    bf16* vtb   = (bf16*)(ws + 92274688);      // 16 MB  V^T [B,H,D,T]

    cvt_f32_bf16<<<4096, 256, 0, stream>>>(x,      xb,    szX);
    cvt_f32_bf16<<<2048, 256, 0, stream>>>(w_qkv,  wqkvb, szWq);
    cvt_f32_bf16<<<1024, 256, 0, stream>>>(w_proj, wprjb, szWp);

    // qkv = x * Wqkv^T : M=8192, N=3072, K=1024
    gemm_wmma<1><<<dim3(3072 / 64, 8192 / 512), 256, 0, stream>>>(
        xb, wqkvb, (void*)qkvb, BB * TT, 3 * CC, CC);

    // V^T: [B,H,D,T]
    transpose_v<<<4096, 256, 0, stream>>>(qkvb, vtb);

    // flash attention: grid (T/128, B*H)
    flash_attn<<<dim3(TT / 128, BB * HH), 256, 0, stream>>>(qkvb, vtb, yb);

    // out = y * Wproj^T : M=8192, N=1024, K=1024 (fp32 output)
    gemm_wmma<0><<<dim3(1024 / 64, 8192 / 512), 256, 0, stream>>>(
        yb, wprjb, d_out, BB * TT, CC, CC);
}
